// NeRFNGPNet_62938450755822
// MI455X (gfx1250) — matrix operand order repaired and checked by reading, not compiled, not run on hardware
//
#include <hip/hip_runtime.h>
#include <hip/hip_bf16.h>

typedef __bf16 bf16_t;
typedef __attribute__((ext_vector_type(16))) __bf16 v16bf;
typedef __attribute__((ext_vector_type(8)))  float  v8f;

#define NPTS      (1u << 20)
#define THREADS   256
#define PTS_BLOCK 256
#define BATCHES   4
#define NBLOCKS   (NPTS / (PTS_BLOCK * BATCHES))   // 1024

#define PLANE_ELEMS (32u * 256u * 256u)            // 2,097,152 floats
#define TP_BYTES    (3ull * PLANE_ELEMS * 4ull)    // 25,165,824 bytes

// ---- LDS layout (units: bf16 elements) ----
#define FSTRIDE   100      // feat row stride: 200B rows (8B aligned), 50 dwords
#define OFF_W1T   0        // 64 rows x 98   (w_sigma1: 96x64)
#define OFF_W2T   6272     // 16 rows x 66   (w_sigma2: 64x16)
#define OFF_WC1T  7328     // 64 rows x 34   (w_color1: 15x64, K padded to 32)
#define OFF_WC2T  9504     // 64 rows x 66   (w_color2: 64x64)
#define OFF_WC3T  13728    // 16 rows x 66   (w_color3: 64x3, N padded to 16)
#define OFF_FEAT  14784    // 8 waves x 32 rows x FSTRIDE
#define OFF_H     (14784 + 8 * 32 * FSTRIDE)       // 8 waves x 32 rows x 66
#define SMEM_BF16 (OFF_H + 8 * 32 * 66)
#define SMEM_BYTES (SMEM_BF16 * 2)

__device__ __forceinline__ int clampi(int v, int lo, int hi) {
    return v < lo ? lo : (v > hi ? hi : v);
}

__device__ __forceinline__ float relu1(float v) {
    // Single v_med3_f32: clamp to [0, FLT_MAX].
    return __builtin_amdgcn_fmed3f(v, 0.0f, 3.402823466e38f);
}

__device__ __forceinline__ void lds_fence() {
    // LDS ops are in-order per wave on CDNA5; this is a compiler reordering
    // barrier plus an explicit DS-counter drain between pipeline phases.
    asm volatile("s_wait_dscnt 0" ::: "memory");
}

__device__ __forceinline__ v8f wmma_bf16(v16bf a, v16bf b, v8f c) {
    return __builtin_amdgcn_wmma_f32_16x16x32_bf16(
        /*neg_a=*/false, a, /*neg_b=*/false, b,
        /*c_mod=*/(short)0, c, /*reuse_a=*/false, /*reuse_b=*/false);
}

// A-fragment (16x32 bf16). Per ISA layout: lanes 0-15 hold K 0-7 & 16-23,
// lanes 16-31 hold K 8-15 & 24-31 (pairs packed per VGPR).
__device__ __forceinline__ v16bf load_a_frag(const bf16_t* __restrict__ row,
                                             int k0, int lane) {
    const int half8 = (lane >> 4) << 3;
    v16bf a;
#pragma unroll
    for (int j = 0; j < 8; ++j) {
        const int kk = k0 + half8 + ((j < 4) ? (j * 2) : (16 + (j - 4) * 2));
        a[2 * j]     = row[kk];
        a[2 * j + 1] = row[kk + 1];
    }
    return a;
}

// B-fragment (32x16 bf16). Weights in LDS transposed wT[n][k]:
// lane n%16 holds column n; lanes 0-15 K 0-7 & 16-23, lanes 16-31 K 8-15 & 24-31.
__device__ __forceinline__ v16bf load_b_frag(const bf16_t* __restrict__ wT,
                                             int stride, int ntile, int k0,
                                             int lane) {
    const bf16_t* p = wT + (ntile * 16 + (lane & 15)) * stride + k0 +
                      ((lane >> 4) << 3);
    v16bf b;
#pragma unroll
    for (int j = 0; j < 16; ++j) b[j] = p[(j < 8) ? j : (j + 8)];
    return b;
}

struct BilinTap {
    int o00, o01, o10, o11;
    float w00, w01, w10, w11;
};

__device__ __forceinline__ BilinTap make_tap(float u01, float v01) {
    BilinTap t;
    const float px = u01 * 255.0f;
    const float py = v01 * 255.0f;
    const float x0f = floorf(px), y0f = floorf(py);
    const float wx = px - x0f, wy = py - y0f;
    const int x0 = clampi((int)x0f, 0, 255);
    const int y0 = clampi((int)y0f, 0, 255);
    const int x1 = (x0 + 1 > 255) ? 255 : x0 + 1;
    const int y1 = (y0 + 1 > 255) ? 255 : y0 + 1;
    t.o00 = y0 * 256 + x0; t.o01 = y0 * 256 + x1;
    t.o10 = y1 * 256 + x0; t.o11 = y1 * 256 + x1;
    t.w00 = (1.f - wx) * (1.f - wy); t.w01 = wx * (1.f - wy);
    t.w10 = (1.f - wx) * wy;         t.w11 = wx * wy;
    return t;
}

// Direct (C,H,W) gather (fallback path).
__device__ __forceinline__ void sample_plane_chw(const float* __restrict__ plane,
                                                 float u01, float v01,
                                                 bf16_t* __restrict__ dst) {
    const BilinTap t = make_tap(u01, v01);
#pragma unroll 8
    for (int c = 0; c < 32; ++c) {
        const float* pc = plane + (c << 16);
        const float v = pc[t.o00] * t.w00 + pc[t.o01] * t.w01 +
                        pc[t.o10] * t.w10 + pc[t.o11] * t.w11;
        dst[c] = (bf16_t)v;
    }
}

// Transposed (H*W, C) gather: each corner = one aligned 128B cacheline.
__device__ __forceinline__ void sample_plane_hwc(const float* __restrict__ pt,
                                                 float u01, float v01,
                                                 bf16_t* __restrict__ dst) {
    const BilinTap t = make_tap(u01, v01);
    const float4* p00 = (const float4*)(pt + (unsigned)t.o00 * 32u);
    const float4* p01 = (const float4*)(pt + (unsigned)t.o01 * 32u);
    const float4* p10 = (const float4*)(pt + (unsigned)t.o10 * 32u);
    const float4* p11 = (const float4*)(pt + (unsigned)t.o11 * 32u);
#pragma unroll
    for (int c4 = 0; c4 < 8; ++c4) {
        const float4 a = p00[c4], b = p01[c4], c = p10[c4], d = p11[c4];
        dst[c4 * 4 + 0] = (bf16_t)(a.x * t.w00 + b.x * t.w01 + c.x * t.w10 + d.x * t.w11);
        dst[c4 * 4 + 1] = (bf16_t)(a.y * t.w00 + b.y * t.w01 + c.y * t.w10 + d.y * t.w11);
        dst[c4 * 4 + 2] = (bf16_t)(a.z * t.w00 + b.z * t.w01 + c.z * t.w10 + d.z * t.w11);
        dst[c4 * 4 + 3] = (bf16_t)(a.w * t.w00 + b.w * t.w01 + c.w * t.w10 + d.w * t.w11);
    }
}

// Repack (32,256,256) planes -> (256*256,32); fully coalesced writes.
extern "C" __global__ __launch_bounds__(256) void transpose_planes(
    const float* __restrict__ p0, const float* __restrict__ p1,
    const float* __restrict__ p2, float* __restrict__ dst)
{
    const unsigned bid = blockIdx.x;           // 3 * 8192 blocks
    const unsigned plane = bid >> 13;
    const unsigned chunk = bid & 8191u;
    const unsigned c = threadIdx.x & 31u;
    const unsigned pos = chunk * 8u + (threadIdx.x >> 5);
    const float* src = (plane == 0) ? p0 : (plane == 1) ? p1 : p2;
    dst[plane * PLANE_ELEMS + pos * 32u + c] = src[c * 65536u + pos];
}

template <bool TP>
__global__ __launch_bounds__(THREADS) void nerf_triplane_wmma(
    const float* __restrict__ x,
    const float* __restrict__ plane_xy,   // TP: (HW,C) repacked, else (C,H,W)
    const float* __restrict__ plane_xz,
    const float* __restrict__ plane_yz,
    const float* __restrict__ center,
    const float* __restrict__ scale,
    const float* __restrict__ w_sigma1,   // (96,64)
    const float* __restrict__ w_sigma2,   // (64,16)
    const float* __restrict__ w_color1,   // (15,64)
    const float* __restrict__ w_color2,   // (64,64)
    const float* __restrict__ w_color3,   // (64,3)
    float* __restrict__ out)              // color (N,3) then sigma (N)
{
    extern __shared__ bf16_t smem[];
    bf16_t* w1T  = smem + OFF_W1T;
    bf16_t* w2T  = smem + OFF_W2T;
    bf16_t* wc1T = smem + OFF_WC1T;
    bf16_t* wc2T = smem + OFF_WC2T;
    bf16_t* wc3T = smem + OFF_WC3T;

    const int tid  = threadIdx.x;
    const int lane = tid & 31;
    const int wave = tid >> 5;

    bf16_t* featW = smem + OFF_FEAT + wave * (32 * FSTRIDE);
    bf16_t* hW    = smem + OFF_H    + wave * (32 * 66);

    // ---- Stage weights once per block: f32 -> bf16, transposed [n][k] ----
    for (int i = tid; i < 96 * 64; i += THREADS) {
        const int k = i >> 6, n = i & 63;
        w1T[n * 98 + k] = (bf16_t)w_sigma1[i];
    }
    for (int i = tid; i < 64 * 16; i += THREADS) {
        const int k = i >> 4, n = i & 15;
        w2T[n * 66 + k] = (bf16_t)w_sigma2[i];
    }
    for (int i = tid; i < 64 * 32; i += THREADS) {      // K padded 15 -> 32
        const int n = i >> 5, k = i & 31;
        wc1T[n * 34 + k] = (k < 15) ? (bf16_t)w_color1[k * 64 + n] : (bf16_t)0.0f;
    }
    for (int i = tid; i < 64 * 64; i += THREADS) {
        const int k = i >> 6, n = i & 63;
        wc2T[n * 66 + k] = (bf16_t)w_color2[i];
    }
    for (int i = tid; i < 16 * 64; i += THREADS) {      // N padded 3 -> 16
        const int n = i >> 6, k = i & 63;
        wc3T[n * 66 + k] = (n < 3) ? (bf16_t)w_color3[k * 3 + n] : (bf16_t)0.0f;
    }
    __syncthreads();

    const float cx = center[0], cy = center[1], cz = center[2];
    const float sx = scale[0],  sy = scale[1],  sz = scale[2];
    const int half8 = (lane >> 4) << 3;
    const int col   = lane & 15;

    // ---- Batch loop: amortize weight staging over 4 x 256 points ----
#pragma unroll 1
    for (int batch = 0; batch < BATCHES; ++batch) {
        const int pblk  = (blockIdx.x * BATCHES + batch) * PTS_BLOCK;
        const int pwave = pblk + wave * 32;

        // Phase 0: triplane bilinear gather, one point per thread.
        {
            const int p = pblk + tid;
            const float* xp = x + 3 * (unsigned)p;
            const float xn = fminf(fmaxf((xp[0] - cx) / sx + 0.5f, 0.f), 1.f);
            const float yn = fminf(fmaxf((xp[1] - cy) / sy + 0.5f, 0.f), 1.f);
            const float zn = fminf(fmaxf((xp[2] - cz) / sz + 0.5f, 0.f), 1.f);
            bf16_t* frow = featW + lane * FSTRIDE;
            if (TP) {
                sample_plane_hwc(plane_xy, xn, yn, frow + 0);
                sample_plane_hwc(plane_xz, xn, zn, frow + 32);
                sample_plane_hwc(plane_yz, yn, zn, frow + 64);
            } else {
                sample_plane_chw(plane_xy, xn, yn, frow + 0);
                sample_plane_chw(plane_xz, xn, zn, frow + 32);
                sample_plane_chw(plane_yz, yn, zn, frow + 64);
            }
        }
        lds_fence();

        // Layer 1: h = relu(feat(96) @ w_sigma1(96x64))
#pragma unroll
        for (int mt = 0; mt < 2; ++mt) {
            const bf16_t* arow = featW + (mt * 16 + col) * FSTRIDE;
            const v16bf a0 = load_a_frag(arow, 0,  lane);
            const v16bf a1 = load_a_frag(arow, 32, lane);
            const v16bf a2 = load_a_frag(arow, 64, lane);
#pragma unroll
            for (int nt = 0; nt < 4; ++nt) {
                v8f acc = {};
                acc = wmma_bf16(a0, load_b_frag(w1T, 98, nt, 0,  lane), acc);
                acc = wmma_bf16(a1, load_b_frag(w1T, 98, nt, 32, lane), acc);
                acc = wmma_bf16(a2, load_b_frag(w1T, 98, nt, 64, lane), acc);
                const int n = nt * 16 + col;
#pragma unroll
                for (int r = 0; r < 8; ++r) {
                    const int m = mt * 16 + half8 + r;
                    hW[m * 66 + n] = (bf16_t)relu1(acc[r]);
                }
            }
        }
        lds_fence();

        // Layer 2: out16 = h(64) @ w_sigma2(64x16); sigma + color input.
#pragma unroll
        for (int mt = 0; mt < 2; ++mt) {
            const bf16_t* arow = hW + (mt * 16 + col) * 66;
            v8f acc = {};
            acc = wmma_bf16(load_a_frag(arow, 0,  lane),
                            load_b_frag(w2T, 66, 0, 0,  lane), acc);
            acc = wmma_bf16(load_a_frag(arow, 32, lane),
                            load_b_frag(w2T, 66, 0, 32, lane), acc);
#pragma unroll
            for (int r = 0; r < 8; ++r) {
                const int m = mt * 16 + half8 + r;
                if (col == 0) {
                    out[3u * NPTS + (unsigned)(pwave + m)] = acc[r];  // sigma
                    featW[m * FSTRIDE + 31] = (bf16_t)0.0f;           // pad K=31
                } else {
                    featW[m * FSTRIDE + (col - 1)] = (bf16_t)acc[r];  // cin K=0..14
                }
                featW[m * FSTRIDE + 15 + col] = (bf16_t)0.0f;         // pad K=15..30
            }
        }
        lds_fence();

        // Color layer 1: c1 = relu(cin(15,pad32) @ w_color1(->64))
#pragma unroll
        for (int mt = 0; mt < 2; ++mt) {
            const bf16_t* arow = featW + (mt * 16 + col) * FSTRIDE;
            const v16bf a0 = load_a_frag(arow, 0, lane);
#pragma unroll
            for (int nt = 0; nt < 4; ++nt) {
                v8f acc = {};
                acc = wmma_bf16(a0, load_b_frag(wc1T, 34, nt, 0, lane), acc);
                const int n = nt * 16 + col;
#pragma unroll
                for (int r = 0; r < 8; ++r) {
                    const int m = mt * 16 + half8 + r;
                    hW[m * 66 + n] = (bf16_t)relu1(acc[r]);
                }
            }
        }
        lds_fence();

        // Color layer 2: c2 = relu(c1(64) @ w_color2(64x64))
#pragma unroll
        for (int mt = 0; mt < 2; ++mt) {
            const bf16_t* arow = hW + (mt * 16 + col) * 66;
            const v16bf a0 = load_a_frag(arow, 0,  lane);
            const v16bf a1 = load_a_frag(arow, 32, lane);
#pragma unroll
            for (int nt = 0; nt < 4; ++nt) {
                v8f acc = {};
                acc = wmma_bf16(a0, load_b_frag(wc2T, 66, nt, 0,  lane), acc);
                acc = wmma_bf16(a1, load_b_frag(wc2T, 66, nt, 32, lane), acc);
                const int n = nt * 16 + col;
#pragma unroll
                for (int r = 0; r < 8; ++r) {
                    const int m = mt * 16 + half8 + r;
                    featW[m * FSTRIDE + n] = (bf16_t)relu1(acc[r]);
                }
            }
        }
        lds_fence();

        // Color layer 3: color = sigmoid(c2(64) @ w_color3(64x3,padN16))
#pragma unroll
        for (int mt = 0; mt < 2; ++mt) {
            const bf16_t* arow = featW + (mt * 16 + col) * FSTRIDE;
            v8f acc = {};
            acc = wmma_bf16(load_a_frag(arow, 0,  lane),
                            load_b_frag(wc3T, 66, 0, 0,  lane), acc);
            acc = wmma_bf16(load_a_frag(arow, 32, lane),
                            load_b_frag(wc3T, 66, 0, 32, lane), acc);
            if (col < 3) {
#pragma unroll
                for (int r = 0; r < 8; ++r) {
                    const int m = mt * 16 + half8 + r;
                    const float e = __expf(-acc[r]);
                    const float s = __builtin_amdgcn_rcpf(1.0f + e);
                    out[(unsigned)(pwave + m) * 3u + col] = s;
                }
            }
        }
        lds_fence();   // scratch reuse across batches
    }
}

extern "C" void kernel_launch(void* const* d_in, const int* in_sizes, int n_in,
                              void* d_out, int out_size, void* d_ws, size_t ws_size,
                              hipStream_t stream) {
    (void)in_sizes; (void)n_in; (void)out_size;
    const float* x        = (const float*)d_in[0];
    // d_in[1] = view directions 'd' — unused by the reference network.
    const float* plane_xy = (const float*)d_in[2];
    const float* plane_xz = (const float*)d_in[3];
    const float* plane_yz = (const float*)d_in[4];
    const float* center   = (const float*)d_in[5];
    const float* scale    = (const float*)d_in[6];
    const float* w_sigma1 = (const float*)d_in[7];
    const float* w_sigma2 = (const float*)d_in[8];
    const float* w_color1 = (const float*)d_in[9];
    const float* w_color2 = (const float*)d_in[10];
    const float* w_color3 = (const float*)d_in[11];
    float* out = (float*)d_out;

    if (ws_size >= TP_BYTES) {
        // Repack planes (C,H,W) -> (HW,C): every bilinear corner becomes one
        // fully-utilized 128B cacheline for random sample positions.
        float* tp = (float*)d_ws;
        transpose_planes<<<3 * 8192, 256, 0, stream>>>(plane_xy, plane_xz,
                                                       plane_yz, tp);
        (void)hipFuncSetAttribute((const void*)nerf_triplane_wmma<true>,
                                  hipFuncAttributeMaxDynamicSharedMemorySize,
                                  SMEM_BYTES);
        nerf_triplane_wmma<true><<<NBLOCKS, THREADS, SMEM_BYTES, stream>>>(
            x, tp, tp + PLANE_ELEMS, tp + 2 * PLANE_ELEMS, center, scale,
            w_sigma1, w_sigma2, w_color1, w_color2, w_color3, out);
    } else {
        (void)hipFuncSetAttribute((const void*)nerf_triplane_wmma<false>,
                                  hipFuncAttributeMaxDynamicSharedMemorySize,
                                  SMEM_BYTES);
        nerf_triplane_wmma<false><<<NBLOCKS, THREADS, SMEM_BYTES, stream>>>(
            x, plane_xy, plane_xz, plane_yz, center, scale,
            w_sigma1, w_sigma2, w_color1, w_color2, w_color3, out);
    }
}
